// DETMDecoder_40535901340195
// MI455X (gfx1250) — compile-verified, matrix operand order checked by reading, not run on hardware
//
#include <hip/hip_runtime.h>
#include <hip/hip_bf16.h>

// Problem constants
#define NB   128            // batch B
#define NK   50             // topics K
#define NKP  64             // padded topics per batch (4 m-tiles of 16)
#define NL   300            // embedding length
#define NLP  320            // padded L (10 k-steps of 32)
#define NKSTEP 10           // NLP / 32
#define NV   50000          // vocab
#define NVP  50048          // vocab padded to multiple of 64 (4 n-tiles)
#define NGROUPS   782       // (NVP/16) / 4  (4 n-tiles per register-blocked group)
#define CHUNK_GROUPS 17     // groups per softmax chunk (782 = 46 * 17)
#define NCHUNK    46        // chunks per row
#define MROWS (NB * NKP)    // 8192 padded rows
#define WAVES_PER_BLK 8
#define NEG_INF (-__builtin_inff())

typedef __attribute__((ext_vector_type(16))) __bf16 v16bf;
typedef __attribute__((ext_vector_type(8)))  float  v8f;

union FragU { v16bf v; uint4 q[2]; };

// ---------- helpers ----------
__device__ __forceinline__ unsigned short f2bf(float f) {
    unsigned u = __float_as_uint(f);
    u = u + 0x7FFFu + ((u >> 16) & 1u);   // round to nearest even
    return (unsigned short)(u >> 16);
}

// Preload the full A panel for 2 M-tiles x K=320 into registers (20 frags = 160 VGPRs).
__device__ __forceinline__ void load_apanel(const unsigned short* __restrict__ Ab,
                                            int mrow, int lane, FragU apan[2][NKSTEP]) {
    const int r  = lane & 15;
    const int hi = lane >> 4;
    const unsigned short* pa0 = Ab + (size_t)(mrow + r) * NLP;
    const unsigned short* pa1 = pa0 + (size_t)16 * NLP;
#pragma unroll
    for (int kk = 0; kk < NKSTEP; ++kk) {
        const int k0 = kk * 32;
        apan[0][kk].q[0] = *reinterpret_cast<const uint4*>(pa0 + k0 + hi * 8);
        apan[0][kk].q[1] = *reinterpret_cast<const uint4*>(pa0 + k0 + 16 + hi * 8);
        apan[1][kk].q[0] = *reinterpret_cast<const uint4*>(pa1 + k0 + hi * 8);
        apan[1][kk].q[1] = *reinterpret_cast<const uint4*>(pa1 + k0 + 16 + hi * 8);
    }
}

// 32x64 macro-tile GEMM with register-resident A panel and explicitly
// double-buffered B fragments: loads for k-step kk+1 are issued before the
// 8 WMMAs of k-step kk, so one WMMA octet covers the L2 latency of the next
// fragment set. Fully unrolled -> parity indexing costs no copies.
__device__ __forceinline__ void gemm_panel(const FragU apan[2][NKSTEP],
                                           const unsigned short* __restrict__ Wb,
                                           int nrow, int lane, v8f acc[2][4]) {
    const int r  = lane & 15;
    const int hi = lane >> 4;
    const unsigned short* pb = Wb + (size_t)(nrow + r) * NLP;
    FragU bb[2][4];
#pragma unroll
    for (int t = 0; t < 4; ++t) {
        bb[0][t].q[0] = *reinterpret_cast<const uint4*>(pb + t * 16 * NLP + hi * 8);
        bb[0][t].q[1] = *reinterpret_cast<const uint4*>(pb + t * 16 * NLP + 16 + hi * 8);
    }
#pragma unroll
    for (int kk = 0; kk < NKSTEP; ++kk) {
        const int cur = kk & 1;
        if (kk < NKSTEP - 1) {
            const int k0n = (kk + 1) * 32;
#pragma unroll
            for (int t = 0; t < 4; ++t) {
                bb[cur ^ 1][t].q[0] = *reinterpret_cast<const uint4*>(pb + t * 16 * NLP + k0n + hi * 8);
                bb[cur ^ 1][t].q[1] = *reinterpret_cast<const uint4*>(pb + t * 16 * NLP + k0n + 16 + hi * 8);
            }
        }
#pragma unroll
        for (int t = 0; t < 4; ++t) {
            acc[0][t] = __builtin_amdgcn_wmma_f32_16x16x32_bf16(false, apan[0][kk].v, false, bb[cur][t].v, (short)0, acc[0][t], false, false);
            acc[1][t] = __builtin_amdgcn_wmma_f32_16x16x32_bf16(false, apan[1][kk].v, false, bb[cur][t].v, (short)0, acc[1][t], false, false);
        }
    }
}

// ---------- pack kernels (f32 -> padded bf16) ----------
__global__ void __launch_bounds__(256) packA(const float* __restrict__ alpha,
                                             unsigned short* __restrict__ Ab) {
    int idx = blockIdx.x * blockDim.x + threadIdx.x;       // MROWS*NLP threads
    if (idx >= MROWS * NLP) return;
    int m = idx / NLP, l = idx % NLP;
    int b = m >> 6, k = m & 63;
    float v = (k < NK && l < NL) ? alpha[((size_t)b * NK + k) * NL + l] : 0.0f;
    Ab[idx] = f2bf(v);
}
__global__ void __launch_bounds__(256) packW(const float* __restrict__ wemb,
                                             unsigned short* __restrict__ Wb) {
    size_t idx = (size_t)blockIdx.x * blockDim.x + threadIdx.x; // NVP*NLP threads
    if (idx >= (size_t)NVP * NLP) return;
    size_t vv = idx / NLP, l = idx % NLP;
    float v = (vv < NV && l < NL) ? wemb[vv * NL + l] : 0.0f;
    Wb[idx] = f2bf(v);
}

// ---------- pass 1: fused online-softmax partials (local max + local sum) ----------
// Each wave owns 32 rows x one chunk of 17 groups (1088 columns); A panel lives
// in registers for the whole chunk, B streams from L2. Running (m, s) per row
// merged online; one (max,sum) partial per (row, chunk). No atomics.
__global__ void __launch_bounds__(256)
__attribute__((amdgpu_waves_per_eu(1)))
pass1_partials(const unsigned short* __restrict__ Ab,
               const unsigned short* __restrict__ Wb,
               float* __restrict__ pmax,
               float* __restrict__ psum) {
    const int wave = threadIdx.x >> 5;
    const int lane = threadIdx.x & 31;
    const int r    = lane & 15;
    const int hi   = lane >> 4;
    const int chunk = blockIdx.x * WAVES_PER_BLK + wave;   // wave-uniform guard
    if (chunk >= NCHUNK) return;
    const int mbase = blockIdx.y * 32;                     // 2 m-tiles

    FragU apan[2][NKSTEP];
    load_apanel(Ab, mbase, lane, apan);

    float mloc[2][8], sloc[2][8];
#pragma unroll
    for (int mt = 0; mt < 2; ++mt)
#pragma unroll
        for (int j = 0; j < 8; ++j) { mloc[mt][j] = NEG_INF; sloc[mt][j] = 0.0f; }

    for (int g = 0; g < CHUNK_GROUPS; ++g) {
        const int nrow = (chunk * CHUNK_GROUPS + g) * 64;
        v8f acc[2][4] = {};
        gemm_panel(apan, Wb, nrow, lane, acc);
#pragma unroll
        for (int mt = 0; mt < 2; ++mt) {
#pragma unroll
            for (int j = 0; j < 8; ++j) {
                // group max over the 4 n-tiles and 16 column-lanes of this row
                float x = NEG_INF;
#pragma unroll
                for (int t = 0; t < 4; ++t) {
                    float xv = (nrow + t * 16 + r < NV) ? acc[mt][t][j] : NEG_INF;
                    x = fmaxf(x, xv);
                }
                for (int s = 1; s < 16; s <<= 1) x = fmaxf(x, __shfl_xor(x, s, 32));
                // group sum of exp(x - groupmax); masked cols give exp(-inf)=0
                float gs = 0.0f;
#pragma unroll
                for (int t = 0; t < 4; ++t) {
                    float xv = (nrow + t * 16 + r < NV) ? acc[mt][t][j] : NEG_INF;
                    gs += __expf(xv - x);
                }
                for (int s = 1; s < 16; s <<= 1) gs += __shfl_xor(gs, s, 32);
                // online merge into running (m, s)
                float nm = fmaxf(mloc[mt][j], x);
                sloc[mt][j] = sloc[mt][j] * __expf(mloc[mt][j] - nm) + gs * __expf(x - nm);
                mloc[mt][j] = nm;
            }
        }
    }
    if ((lane & 15) == 0) {
#pragma unroll
        for (int mt = 0; mt < 2; ++mt)
#pragma unroll
            for (int j = 0; j < 8; ++j) {
                int m = mbase + mt * 16 + j + 8 * hi;
                pmax[(size_t)chunk * MROWS + m] = mloc[mt][j];
                psum[(size_t)chunk * MROWS + m] = sloc[mt][j];
            }
    }
}

// ---------- combine: global rowmax + coef = theta / Z ----------
__global__ void __launch_bounds__(256) combineK(const float* __restrict__ theta,
                                                const float* __restrict__ pmax,
                                                const float* __restrict__ psum,
                                                float* __restrict__ rowmax,
                                                float* __restrict__ coef) {
    int m = blockIdx.x * blockDim.x + threadIdx.x;
    if (m >= MROWS) return;
    float gm = NEG_INF;
    for (int c = 0; c < NCHUNK; ++c) gm = fmaxf(gm, pmax[(size_t)c * MROWS + m]);
    float Z = 0.0f;
    for (int c = 0; c < NCHUNK; ++c)
        Z += psum[(size_t)c * MROWS + m] * __expf(pmax[(size_t)c * MROWS + m] - gm);
    rowmax[m] = gm;
    int b = m >> 6, k = m & 63;
    float t = (k < NK) ? theta[b * NK + k] : 0.0f;
    coef[m] = t / Z;   // padded rows: t=0, Z=50000 -> coef 0
}

// ---------- pass 2: out[b,v] = sum_k coef * exp(logit - rowmax), direct store ----------
__global__ void __launch_bounds__(256)
__attribute__((amdgpu_waves_per_eu(1)))
pass2_out(const unsigned short* __restrict__ Ab,
          const unsigned short* __restrict__ Wb,
          const float* __restrict__ rowmax,
          const float* __restrict__ coef,
          float* __restrict__ out) {
    const int wave = threadIdx.x >> 5;
    const int lane = threadIdx.x & 31;
    const int r    = lane & 15;
    const int hi   = lane >> 4;
    const int ng = blockIdx.x * WAVES_PER_BLK + wave;      // wave-uniform guard
    if (ng >= NGROUPS) return;
    const int b = blockIdx.y;
    const int nrow = ng * 64;

    float col[4] = {0.0f, 0.0f, 0.0f, 0.0f};
#pragma unroll
    for (int mt2 = 0; mt2 < 2; ++mt2) {                    // 64 padded topic rows
        const int mrow = b * NKP + mt2 * 32;
        FragU apan[2][NKSTEP];
        load_apanel(Ab, mrow, lane, apan);
        v8f acc[2][4] = {};
        gemm_panel(apan, Wb, nrow, lane, acc);
#pragma unroll
        for (int mt = 0; mt < 2; ++mt)
#pragma unroll
            for (int j = 0; j < 8; ++j) {
                int m = mrow + mt * 16 + j + 8 * hi;
                float cf = coef[m];
                float rm = rowmax[m];
#pragma unroll
                for (int t = 0; t < 4; ++t)
                    col[t] += cf * __expf(acc[mt][t][j] - rm);
            }
    }
#pragma unroll
    for (int t = 0; t < 4; ++t) {
        float c2 = col[t] + __shfl_xor(col[t], 16, 32);    // fold the two M halves
        int v = nrow + t * 16 + r;
        if (lane < 16 && v < NV) out[(size_t)b * NV + v] = c2;
    }
}

extern "C" void kernel_launch(void* const* d_in, const int* in_sizes, int n_in,
                              void* d_out, int out_size, void* d_ws, size_t ws_size,
                              hipStream_t stream) {
    (void)in_sizes; (void)n_in; (void)out_size; (void)ws_size;
    const float* theta = (const float*)d_in[0];   // (B,K)
    const float* alpha = (const float*)d_in[1];   // (B,K,L)
    const float* wemb  = (const float*)d_in[2];   // (V,L)
    float* out = (float*)d_out;                   // (B,V)

    char* ws = (char*)d_ws;
    size_t off = 0;
    auto carve = [&](size_t bytes) -> char* {
        char* p = ws + off;
        off += (bytes + 255) & ~(size_t)255;
        return p;
    };
    unsigned short* Ab     = (unsigned short*)carve((size_t)MROWS * NLP * 2); //  5.2 MB
    unsigned short* Wb     = (unsigned short*)carve((size_t)NVP * NLP * 2);   // 32.0 MB
    float*          pmax   = (float*)carve((size_t)NCHUNK * MROWS * 4);       //  1.5 MB
    float*          psum   = (float*)carve((size_t)NCHUNK * MROWS * 4);       //  1.5 MB
    float*          rowmax = (float*)carve((size_t)MROWS * 4);
    float*          coef   = (float*)carve((size_t)MROWS * 4);

    const int TPB = 256;
    packA<<<(MROWS * NLP + TPB - 1) / TPB, TPB, 0, stream>>>(alpha, Ab);
    packW<<<(int)(((size_t)NVP * NLP + TPB - 1) / TPB), TPB, 0, stream>>>(wemb, Wb);

    dim3 g1((NCHUNK + WAVES_PER_BLK - 1) / WAVES_PER_BLK, MROWS / 32);  // (6, 256)
    pass1_partials<<<g1, TPB, 0, stream>>>(Ab, Wb, pmax, psum);

    combineK<<<(MROWS + TPB - 1) / TPB, TPB, 0, stream>>>(theta, pmax, psum, rowmax, coef);

    dim3 g2((NGROUPS + WAVES_PER_BLK - 1) / WAVES_PER_BLK, NB);         // (98, 128)
    pass2_out<<<g2, TPB, 0, stream>>>(Ab, Wb, rowmax, coef, out);
}
